// MS_MoE_Conv_53472342835682
// MI455X (gfx1250) — matrix operand order repaired and checked by reading, not compiled, not run on hardware
//
#include <hip/hip_runtime.h>
#include <math.h>

// ---------------------------------------------------------------------------
// MS-MoE spiking conv block for gfx1250 (MI455X).
// Heavy path: per-token 64x1024x64 GEMMs on binary spikes via
// V_WMMA_F32_16X16X32_F16 (f16 operands exact for {0,1} spikes, f32 accum).
// Weights staged in LDS; A-tiles preloaded to registers so WMMAs issue
// back-to-back; x tile moved global->LDS via async-LDS loads when available.
// ---------------------------------------------------------------------------

typedef __attribute__((ext_vector_type(16))) _Float16 v16h;
typedef __attribute__((ext_vector_type(8)))  _Float16 v8h;
typedef __attribute__((ext_vector_type(8)))  float    v8f;
typedef __attribute__((ext_vector_type(4)))  float    v4f;

#define Bdim 64
#define Cdim 64
#define HW   1024
#define Ntok 256
#define NEXP 8

#define XF_STRIDE 68   // f32 per spatial row: 16B-aligned 8-float chunks, low conflicts
#define S_STRIDE  72   // f16 per spatial row: 16B-aligned, conflict-free B gathers
#define W_STRIDE  72   // f16 per weight row: conflict-free A gathers

#if defined(__HIP_DEVICE_COMPILE__) && __has_builtin(__builtin_amdgcn_global_load_async_to_lds_b32)
#define ASYNC_X_LOAD 1
typedef __attribute__((address_space(1))) int* as1_i32p;
typedef __attribute__((address_space(3))) int* as3_i32p;
#endif

__device__ __forceinline__ v16h cat8(v8h lo, v8h hi) {
  return __builtin_shufflevector(lo, hi, 0,1,2,3,4,5,6,7,8,9,10,11,12,13,14,15);
}

// ---------------------------------------------------------------------------
// Kernel 0: fold BN (gamma/sqrt(1+eps)) into conv weights (f16) and biases.
// ---------------------------------------------------------------------------
__global__ void prep_weights(const float* __restrict__ w1, const float* __restrict__ b1,
                             const float* __restrict__ g1, const float* __restrict__ be1,
                             const float* __restrict__ w2, const float* __restrict__ b2,
                             const float* __restrict__ g2, const float* __restrict__ be2,
                             _Float16* __restrict__ w1s, _Float16* __restrict__ w2s,
                             float* __restrict__ b1s, float* __restrict__ b2s) {
  int i = blockIdx.x * 256 + threadIdx.x;      // 0 .. 8*64*64-1
  if (i >= NEXP * Cdim * Cdim) return;
  float inv = rsqrtf(1.0f + 1e-5f);
  int eo = i >> 6;                              // e*64 + o
  float s1 = g1[eo] * inv, s2 = g2[eo] * inv;
  w1s[i] = (_Float16)(s1 * w1[i]);
  w2s[i] = (_Float16)(s2 * w2[i]);
  if ((i & 63) == 0) {
    b1s[eo] = s1 * b1[eo] + be1[eo];
    b2s[eo] = s2 * b2[eo] + be2[eo];
  }
}

// ---------------------------------------------------------------------------
// Kernel 1: router LIF over T, spatial mean of spikes per (t,b,c).
// One block per (b,c): 256 threads x 4 consecutive positions (b128 loads).
// ---------------------------------------------------------------------------
__global__ void router_spikes(const float* __restrict__ x, float* __restrict__ spkmean) {
  int bc = blockIdx.x;                 // b*64 + c
  int tid = threadIdx.x;
  const size_t tstride = (size_t)Bdim * Cdim * HW;
  const float* xp = x + (size_t)bc * HW + tid * 4;
  v4f X0 = *(const v4f*)(xp);
  v4f X1 = *(const v4f*)(xp + tstride);
  v4f X2 = *(const v4f*)(xp + 2 * tstride);
  v4f X3 = *(const v4f*)(xp + 3 * tstride);
  float c0 = 0.f, c1 = 0.f, c2 = 0.f, c3 = 0.f;
  #pragma unroll
  for (int j = 0; j < 4; ++j) {
    float v = 0.f, s;
    v = (v + X0[j]) * 0.5f; s = (v >= 1.f) ? 1.f : 0.f; c0 += s; v *= (1.f - s);
    v = (v + X1[j]) * 0.5f; s = (v >= 1.f) ? 1.f : 0.f; c1 += s; v *= (1.f - s);
    v = (v + X2[j]) * 0.5f; s = (v >= 1.f) ? 1.f : 0.f; c2 += s; v *= (1.f - s);
    v = (v + X3[j]) * 0.5f; s = (v >= 1.f) ? 1.f : 0.f; c3 += s;
  }
  __shared__ float red[4][256];
  red[0][tid] = c0; red[1][tid] = c1; red[2][tid] = c2; red[3][tid] = c3;
  __syncthreads();
  for (int s = 128; s > 0; s >>= 1) {
    if (tid < s) {
      #pragma unroll
      for (int ts = 0; ts < 4; ++ts) red[ts][tid] += red[ts][tid + s];
    }
    __syncthreads();
  }
  if (tid < 4) spkmean[tid * (Bdim * Cdim) + bc] = red[tid][0] * (1.0f / (float)HW);
}

// ---------------------------------------------------------------------------
// Kernel 2: logits -> softmax -> top-2 -> normalized dense combine weights.
// ---------------------------------------------------------------------------
__global__ void router_combine(const float* __restrict__ spkmean,
                               const float* __restrict__ rw, const float* __restrict__ rb,
                               const float* __restrict__ rg, const float* __restrict__ rbe,
                               float* __restrict__ combine) {
  int n = threadIdx.x;                               // 0..255 ; n = t*B + b
  const float* sm = spkmean + (n / Bdim) * (Bdim * Cdim) + (n % Bdim) * Cdim;
  float inv = rsqrtf(1.0f + 1e-5f);
  float lg[NEXP];
  float mx = -1e30f;
  #pragma unroll
  for (int e = 0; e < NEXP; ++e) {
    float d = 0.f;
    for (int c = 0; c < Cdim; ++c) d += rw[e * Cdim + c] * sm[c];
    lg[e] = rg[e] * inv * (d + rb[e]) + rbe[e];
    mx = fmaxf(mx, lg[e]);
  }
  float se = 0.f, p[NEXP];
  #pragma unroll
  for (int e = 0; e < NEXP; ++e) { p[e] = __expf(lg[e] - mx); se += p[e]; }
  #pragma unroll
  for (int e = 0; e < NEXP; ++e) p[e] /= se;
  int i1 = 0;
  #pragma unroll
  for (int e = 1; e < NEXP; ++e) if (p[e] > p[i1]) i1 = e;
  int i2 = (i1 == 0) ? 1 : 0;
  #pragma unroll
  for (int e = 0; e < NEXP; ++e) if (e != i1 && e != i2 && p[e] > p[i2]) i2 = e;
  float s = p[i1] + p[i2];
  #pragma unroll
  for (int e = 0; e < NEXP; ++e) combine[n * NEXP + e] = 0.f;
  combine[n * NEXP + i1] = p[i1] / s;
  combine[n * NEXP + i2] = p[i2] / s;
}

// ---------------------------------------------------------------------------
// Stage one 64x64 f16 weight matrix into LDS (padded rows), 256 threads.
// ---------------------------------------------------------------------------
__device__ __forceinline__ void stage_weights(const _Float16* __restrict__ wsrc,
                                              _Float16* wl, int tid) {
  int m = tid >> 2, k0 = (tid & 3) * 16;
  v8h a = *(const v8h*)(wsrc + m * 64 + k0);
  v8h b = *(const v8h*)(wsrc + m * 64 + k0 + 8);
  *(v8h*)(&wl[m * W_STRIDE + k0]) = a;
  *(v8h*)(&wl[m * W_STRIDE + k0 + 8]) = b;
}

// ---------------------------------------------------------------------------
// Wave GEMM: D[64 x 16cols] = W[64x64] * S[64x16]. All 8 A-tiles preloaded
// to registers, then 8 back-to-back WMMAs.
// wl: lane's A base in LDS weights = wB + l16*W_STRIDE + half*16
// bp: lane's B base in LDS spikes  = sT + col*S_STRIDE + half*16
// ---------------------------------------------------------------------------
__device__ __forceinline__ void gemm64(const _Float16* wl, const _Float16* bp,
                                       v8f cacc[4]) {
  // B (32x16 f16): lanes 0-15 -> K 0..15, lanes 16-31 -> K 16..31.
  v16h B0 = cat8(*(const v8h*)(bp),      *(const v8h*)(bp + 8));    // K 0..31
  v16h B1 = cat8(*(const v8h*)(bp + 32), *(const v8h*)(bp + 40));   // K 32..63
  // A (16x32 f16): lanes 0-15 -> K 0..15, lanes 16-31 -> K 16..31.
  v16h A[4][2];
  #pragma unroll
  for (int mt = 0; mt < 4; ++mt)
    #pragma unroll
    for (int kt = 0; kt < 2; ++kt) {
      const _Float16* q = wl + mt * (16 * W_STRIDE) + kt * 32;
      A[mt][kt] = cat8(*(const v8h*)(q), *(const v8h*)(q + 8));
    }
  #pragma unroll
  for (int mt = 0; mt < 4; ++mt) {
    v8f c = {};
    c = __builtin_amdgcn_wmma_f32_16x16x32_f16(false, A[mt][0], false, B0, (short)0, c, false, false);
    c = __builtin_amdgcn_wmma_f32_16x16x32_f16(false, A[mt][1], false, B1, (short)0, c, false, false);
    cacc[mt] = c;
  }
}

// ---------------------------------------------------------------------------
// Kernel 3: experts. Block = (token, 128-col spatial tile), 8 waves.
// Wave w owns output columns [16w, 16w+16).
// out = 2*x + sum_e combine[n,e] * (h_e + y_e)
// ---------------------------------------------------------------------------
__launch_bounds__(256)
__global__ void experts_kernel(const float* __restrict__ x, const float* __restrict__ combine,
                               const _Float16* __restrict__ w1s, const float* __restrict__ b1s,
                               const _Float16* __restrict__ w2s, const float* __restrict__ b2s,
                               float* __restrict__ out) {
  __shared__ float    xf[128 * XF_STRIDE];   // x transposed: [p][c], f32   (34816 B)
  __shared__ _Float16 sT[128 * S_STRIDE];    // spikes:       [p][c], f16   (18432 B)
  __shared__ _Float16 wB[64 * W_STRIDE];     // current layer weights       ( 9216 B)

  int n    = blockIdx.x >> 3;                // token
  int hw0  = (blockIdx.x & 7) * 128;         // spatial tile base
  int tid  = threadIdx.x;
  const float* xn = x + (size_t)n * (Cdim * HW) + hw0;

  // cooperative transposed load of x tile: global -> LDS
  #pragma unroll
  for (int i = 0; i < 32; ++i) {
    int idx = i * 256 + tid;                 // 0..8191
    int c = idx >> 7, p = idx & 127;
#ifdef ASYNC_X_LOAD
    __builtin_amdgcn_global_load_async_to_lds_b32(
        (as1_i32p)(xn + (size_t)c * HW + p),
        (as3_i32p)&xf[p * XF_STRIDE + c], 0, 0);
#else
    xf[p * XF_STRIDE + c] = xn[(size_t)c * HW + p];
#endif
  }
#ifdef ASYNC_X_LOAD
#if __has_builtin(__builtin_amdgcn_s_wait_asynccnt)
  __builtin_amdgcn_s_wait_asynccnt(0);
#else
  asm volatile("s_wait_asynccnt 0" ::: "memory");
#endif
#endif
  __syncthreads();

  int wave = tid >> 5;
  int lane = tid & 31;
  int half = lane >> 4;                      // WMMA K/M-half selector
  int l16  = lane & 15;
  int col  = wave * 16 + l16;                // C/D-layout column (spatial pos)

  // acc = 2*x at C/D-layout positions: (m, col), m = mt*16 + v + half*8
  v8f acc[4];
  #pragma unroll
  for (int mt = 0; mt < 4; ++mt) {
    const float* xr = &xf[col * XF_STRIDE + mt * 16 + half * 8];
    v4f a = *(const v4f*)(xr), b = *(const v4f*)(xr + 4);
    #pragma unroll
    for (int v = 0; v < 4; ++v) { acc[mt][v] = 2.0f * a[v]; acc[mt][v + 4] = 2.0f * b[v]; }
  }

  for (int e = 0; e < NEXP; ++e) {
    float wc = combine[n * NEXP + e];        // uniform per block
    if (wc == 0.0f) continue;                // uniform branch: skip unselected
    float tau = 1.5f + 2.5f * (float)e * (1.0f / 7.0f);

    // stage W1 into LDS; s1 = (x >= tau) into this wave's own 16 columns
    stage_weights(w1s + e * 4096, wB, tid);
    {
      int p  = wave * 16 + (lane >> 1);
      int c0 = (lane & 1) * 32;
      const float* xr = &xf[p * XF_STRIDE + c0];
      _Float16* sw = &sT[p * S_STRIDE + c0];
      #pragma unroll
      for (int j = 0; j < 4; ++j) {
        v4f a = *(const v4f*)(xr + j * 8), b = *(const v4f*)(xr + j * 8 + 4);
        v8h s;
        #pragma unroll
        for (int k = 0; k < 4; ++k) {
          s[k]     = (_Float16)((a[k] >= tau) ? 1.0f : 0.0f);
          s[k + 4] = (_Float16)((b[k] >= tau) ? 1.0f : 0.0f);
        }
        *(v8h*)(sw + j * 8) = s;
      }
    }
    __syncthreads();

    const _Float16* bp = &sT[col * S_STRIDE + half * 16];
    const _Float16* wl = &wB[l16 * W_STRIDE + half * 16];

    // ---- layer 1: h = W1s * s1 + b1s ----
    v8f hreg[4];
    gemm64(wl, bp, hreg);
    #pragma unroll
    for (int mt = 0; mt < 4; ++mt) {
      v8f bb = *(const v8f*)(b1s + e * 64 + mt * 16 + half * 8);
      #pragma unroll
      for (int v = 0; v < 8; ++v) hreg[mt][v] += bb[v];
    }
    __syncthreads();   // everyone done reading wB (W1) and sT (s1)

    // stage W2; s2 = (x + h >= tau) packed v8h stores to own column
    stage_weights(w2s + e * 4096, wB, tid);
    #pragma unroll
    for (int mt = 0; mt < 4; ++mt) {
      const float* xr = &xf[col * XF_STRIDE + mt * 16 + half * 8];
      v4f a = *(const v4f*)(xr), b = *(const v4f*)(xr + 4);
      v8h s;
      #pragma unroll
      for (int v = 0; v < 4; ++v) {
        s[v]     = (_Float16)(((a[v] + hreg[mt][v])     >= tau) ? 1.0f : 0.0f);
        s[v + 4] = (_Float16)(((b[v] + hreg[mt][v + 4]) >= tau) ? 1.0f : 0.0f);
      }
      *(v8h*)(&sT[col * S_STRIDE + mt * 16 + half * 8]) = s;
    }
    __syncthreads();

    // ---- layer 2: y = W2s * s2 + b2s ; acc += wc * (h + y) ----
    v8f yreg[4];
    gemm64(wl, bp, yreg);
    #pragma unroll
    for (int mt = 0; mt < 4; ++mt) {
      v8f bb = *(const v8f*)(b2s + e * 64 + mt * 16 + half * 8);
      #pragma unroll
      for (int v = 0; v < 8; ++v)
        acc[mt][v] += wc * (hreg[mt][v] + yreg[mt][v] + bb[v]);
    }
    __syncthreads();   // protect wB/sT before next expert's staging
  }

  // store: out[n][m][hw0 + col]
  float* on = out + (size_t)n * (Cdim * HW) + hw0;
  #pragma unroll
  for (int mt = 0; mt < 4; ++mt)
    #pragma unroll
    for (int v = 0; v < 8; ++v) {
      int m = mt * 16 + v + half * 8;
      on[(size_t)m * HW + col] = acc[mt][v];
    }
}

// ---------------------------------------------------------------------------
// Host launcher
// ---------------------------------------------------------------------------
extern "C" void kernel_launch(void* const* d_in, const int* in_sizes, int n_in,
                              void* d_out, int out_size, void* d_ws, size_t ws_size,
                              hipStream_t stream) {
  const float* x   = (const float*)d_in[0];
  const float* w1  = (const float*)d_in[1];
  const float* b1  = (const float*)d_in[2];
  const float* g1  = (const float*)d_in[3];
  const float* be1 = (const float*)d_in[4];
  const float* w2  = (const float*)d_in[5];
  const float* b2  = (const float*)d_in[6];
  const float* g2  = (const float*)d_in[7];
  const float* be2 = (const float*)d_in[8];
  const float* rw  = (const float*)d_in[9];
  const float* rb  = (const float*)d_in[10];
  const float* rg  = (const float*)d_in[11];
  const float* rbe = (const float*)d_in[12];
  float* out = (float*)d_out;

  char* ws = (char*)d_ws;
  float*    combine = (float*)(ws);                       //   8 KB (256*8 f32)
  float*    spkmean = (float*)(ws + 8192);                //  64 KB (4*64*64 f32)
  _Float16* w1s     = (_Float16*)(ws + 8192 + 65536);     //  64 KB
  _Float16* w2s     = (_Float16*)(ws + 8192 + 131072);    //  64 KB
  float*    b1s     = (float*)(ws + 8192 + 196608);       //   2 KB
  float*    b2s     = b1s + 512;                          //   2 KB

  prep_weights<<<128, 256, 0, stream>>>(w1, b1, g1, be1, w2, b2, g2, be2,
                                        w1s, w2s, b1s, b2s);
  router_spikes<<<Bdim * Cdim, 256, 0, stream>>>(x, spkmean);
  router_combine<<<1, 256, 0, stream>>>(spkmean, rw, rb, rg, rbe, combine);
  experts_kernel<<<Ntok * 8, 256, 0, stream>>>(x, combine, w1s, b1s, w2s, b2s, out);
}
